// Ext_entLayer_4561255268668
// MI455X (gfx1250) — matrix operand order repaired and checked by reading, not compiled, not run on hardware
//
#include <hip/hip_runtime.h>
#include <hip/hip_bf16.h>

typedef __attribute__((ext_vector_type(16))) __bf16 v16bf;
typedef __attribute__((ext_vector_type(8)))  float  v8f;

#define NNODES 50000
#define NEDGES 500000
#define NRELS  500
#define NTIMES 365
#define ENT    128
#define RELD   128
#define TIMD   64
#define COMP   320   // RELD + ENT + TIMD

typedef unsigned short ushort_t;

// ---------------------------------------------------------------- helpers
__device__ __forceinline__ unsigned pack2_bf16(float a, float b) {
  unsigned ua = __float_as_uint(a);
  unsigned ub = __float_as_uint(b);
  ua += 0x7FFFu + ((ua >> 16) & 1u);   // round-to-nearest-even
  ub += 0x7FFFu + ((ub >> 16) & 1u);
  return (ua >> 16) | (ub & 0xFFFF0000u);
}

union Frag16 { v16bf v; unsigned u[8]; };

__device__ __forceinline__ const ushort_t* seg_ptr(const ushort_t* prel,
                                                   const ushort_t* pent,
                                                   const ushort_t* ptim, int kk) {
  if (kk < RELD)            return prel + kk;
  if (kk < RELD + ENT)      return pent + (kk - RELD);
  return ptim + (kk - RELD - ENT);
}

// A fragment: two 16B loads of 8 bf16 each (no conversion in the hot loop)
__device__ __forceinline__ void loadA_bf16(const ushort_t* prel, const ushort_t* pent,
                                           const ushort_t* ptim, int k0, int k1,
                                           Frag16& a) {
  uint4 q0 = *(const uint4*)seg_ptr(prel, pent, ptim, k0);
  uint4 q1 = *(const uint4*)seg_ptr(prel, pent, ptim, k1);
  a.u[0] = q0.x; a.u[1] = q0.y; a.u[2] = q0.z; a.u[3] = q0.w;
  a.u[4] = q1.x; a.u[5] = q1.y; a.u[6] = q1.z; a.u[7] = q1.w;
}

// B fragment: 16 contiguous bf16 per lane from LDS
__device__ __forceinline__ void loadB_lds(const ushort_t* __restrict__ sW,
                                          int off, Frag16& b) {
  const uint4* p = (const uint4*)(&sW[off]);
  uint4 q0 = p[0], q1 = p[1];
  b.u[0] = q0.x; b.u[1] = q0.y; b.u[2] = q0.z; b.u[3] = q0.w;
  b.u[4] = q1.x; b.u[5] = q1.y; b.u[6] = q1.z; b.u[7] = q1.w;
}

// ---------------------------------------------------------------- kernels
__global__ void zero_ws_kernel(float* __restrict__ p, int n) {
  int i = blockIdx.x * blockDim.x + threadIdx.x;
  if (i < n) p[i] = 0.0f;
}

// one-time f32 -> bf16 table conversion (4 elements / thread)
__global__ void cvt_f32_bf16_kernel(const float* __restrict__ src,
                                    ushort_t* __restrict__ dst, int n4) {
  int i = blockIdx.x * blockDim.x + threadIdx.x;
  if (i >= n4) return;
  float4 f = ((const float4*)src)[i];
  uint2 o;
  o.x = pack2_bf16(f.x, f.y);
  o.y = pack2_bf16(f.z, f.w);
  ((uint2*)dst)[i] = o;
}

// One wave32 per 16-edge tile; masked dual-weight WMMA; atomic scatter-sum.
// A gathered straight as bf16; B double-buffered from LDS-resident bf16 weights.
__global__ void __launch_bounds__(256)
edge_msg_kernel(const ushort_t* __restrict__ rel16,
                const ushort_t* __restrict__ ent16,
                const ushort_t* __restrict__ tim16,
                const ushort_t* __restrict__ wI16,
                const ushort_t* __restrict__ wO16,
                const float* __restrict__ b_I, const float* __restrict__ b_O,
                const int* __restrict__ src, const int* __restrict__ dst,
                const int* __restrict__ b_rel, const int* __restrict__ time_idx,
                const int* __restrict__ inv,
                float* __restrict__ sums, float* __restrict__ cnt) {
  extern __shared__ uint4 smem4[];
  ushort_t* sWI = (ushort_t*)smem4;         // [ENT][COMP] bf16
  ushort_t* sWO = sWI + ENT * COMP;

  {  // stage pre-converted bf16 weights: pure b128 copies
    const uint4* gI = (const uint4*)wI16;
    const uint4* gO = (const uint4*)wO16;
    uint4* sI = smem4;
    uint4* sO = smem4 + (ENT * COMP / 8);
    for (int i = threadIdx.x; i < ENT * COMP / 8; i += blockDim.x) {
      sI[i] = gI[i];
      sO[i] = gO[i];
    }
  }
  __syncthreads();

  const int wave = threadIdx.x >> 5;
  const int lane = threadIdx.x & 31;
  const int tile = blockIdx.x * 8 + wave;
  const int NT   = NEDGES / 16;  // 31250 exactly
  if (tile >= NT) return;

  const int row  = lane & 15;    // A row M / B,C column N
  const int half = lane >> 4;    // lane-half K split
  const int e    = tile * 16 + row;

  const ushort_t* prel = rel16 + (size_t)b_rel[e]    * RELD;
  const ushort_t* pent = ent16 + (size_t)src[e]      * ENT;
  const ushort_t* ptim = tim16 + (size_t)time_idx[e] * TIMD;
  const bool isO = (inv[e] != 0);

  v8f zf = {};
  v8f acc[8];
#pragma unroll
  for (int i = 0; i < 8; ++i) acc[i] = zf;

  Frag16 a;
  loadA_bf16(prel, pent, ptim, half * 16, half * 16 + 8, a);

#pragma unroll
  for (int ks = 0; ks < COMP / 32; ++ks) {   // 10 K-steps of 32
    const int kb = ks * 32;
    Frag16 an;
    if (ks < COMP / 32 - 1)                  // prefetch next A while WMMAs run
      loadA_bf16(prel, pent, ptim, kb + 32 + half * 16, kb + 40 + half * 16, an);

    Frag16 aI, aO;
#pragma unroll
    for (int j = 0; j < 8; ++j) {            // exact per-edge row mask
      aI.u[j] = isO ? 0u : a.u[j];
      aO.u[j] = isO ? a.u[j] : 0u;
    }

    const int koff = kb + half * 16;
    Frag16 bI, bO;                           // B double-buffer
    loadB_lds(sWI, row * COMP + koff, bI);
    loadB_lds(sWO, row * COMP + koff, bO);
#pragma unroll
    for (int nc = 0; nc < 8; ++nc) {
      Frag16 bI2, bO2;
      if (nc < 7) {
        const int n2 = (nc + 1) * 16 + row;
        loadB_lds(sWI, n2 * COMP + koff, bI2);
        loadB_lds(sWO, n2 * COMP + koff, bO2);
      }
      acc[nc] = __builtin_amdgcn_wmma_f32_16x16x32_bf16(
          false, aI.v, false, bI.v, (short)0, acc[nc], false, false);
      acc[nc] = __builtin_amdgcn_wmma_f32_16x16x32_bf16(
          false, aO.v, false, bO.v, (short)0, acc[nc], false, false);
      if (nc < 7) { bI = bI2; bO = bO2; }
    }
    if (ks < COMP / 32 - 1) a = an;
  }

  // epilogue: C element (m,n): m = r + half*8, n = nc*16 + row
  int d8[8], iv8[8];
#pragma unroll
  for (int r = 0; r < 8; ++r) {
    const int em = tile * 16 + half * 8 + r;
    d8[r]  = dst[em];
    iv8[r] = inv[em];
  }
#pragma unroll
  for (int nc = 0; nc < 8; ++nc) {
    const int n = nc * 16 + row;
    const float bi = b_I[n], bo = b_O[n];
#pragma unroll
    for (int r = 0; r < 8; ++r) {
      const float v = acc[nc][r] + (iv8[r] ? bo : bi);
      atomicAdd(&sums[(size_t)d8[r] * ENT + n], v);
    }
  }
  if (half == 0) atomicAdd(&cnt[dst[e]], 1.0f);   // one count per edge
}

// ent_new = ent_emb @ W_S.T + b_S + sums/max(cnt,1)   (WMMA, bf16 W_S in LDS)
__global__ void __launch_bounds__(256)
node_update_kernel(const float* __restrict__ ent_emb,   // fp32 (unused for A)
                   const ushort_t* __restrict__ ent16,
                   const ushort_t* __restrict__ wS16,
                   const float* __restrict__ b_S,
                   const float* __restrict__ sums, const float* __restrict__ cnt,
                   float* __restrict__ out) {
  __shared__ uint4 sWSq[ENT * ENT / 8];      // 32 KB bf16
  {
    const uint4* g = (const uint4*)wS16;
    for (int i = threadIdx.x; i < ENT * ENT / 8; i += blockDim.x) sWSq[i] = g[i];
  }
  __syncthreads();
  const ushort_t* sWS = (const ushort_t*)sWSq;

  const int wave = threadIdx.x >> 5;
  const int lane = threadIdx.x & 31;
  const int tile = blockIdx.x * 8 + wave;
  const int NT   = NNODES / 16;  // 3125 exactly
  if (tile >= NT) return;

  const int row  = lane & 15;
  const int half = lane >> 4;
  const ushort_t* pa = ent16 + (size_t)(tile * 16 + row) * ENT;

  v8f zf = {};
  v8f acc[8];
#pragma unroll
  for (int i = 0; i < 8; ++i) acc[i] = zf;

#pragma unroll
  for (int ks = 0; ks < ENT / 32; ++ks) {    // 4 K-steps
    const int kb = ks * 32;
    Frag16 a;
    {
      uint4 q0 = *(const uint4*)(pa + kb + half * 16);
      uint4 q1 = *(const uint4*)(pa + kb + half * 16 + 8);
      a.u[0] = q0.x; a.u[1] = q0.y; a.u[2] = q0.z; a.u[3] = q0.w;
      a.u[4] = q1.x; a.u[5] = q1.y; a.u[6] = q1.z; a.u[7] = q1.w;
    }
    const int koff = kb + half * 16;
    Frag16 b0;
    loadB_lds(sWS, row * ENT + koff, b0);
#pragma unroll
    for (int nc = 0; nc < 8; ++nc) {
      Frag16 b1;
      if (nc < 7) loadB_lds(sWS, ((nc + 1) * 16 + row) * ENT + koff, b1);
      acc[nc] = __builtin_amdgcn_wmma_f32_16x16x32_bf16(
          false, a.v, false, b0.v, (short)0, acc[nc], false, false);
      if (nc < 7) b0 = b1;
    }
  }

  int   nd8[8];
  float rc8[8];
#pragma unroll
  for (int r = 0; r < 8; ++r) {
    nd8[r] = tile * 16 + half * 8 + r;
    rc8[r] = 1.0f / fmaxf(cnt[nd8[r]], 1.0f);
  }
#pragma unroll
  for (int nc = 0; nc < 8; ++nc) {
    const int n = nc * 16 + row;
    const float bs = b_S[n];
#pragma unroll
    for (int r = 0; r < 8; ++r) {
      const size_t o = (size_t)nd8[r] * ENT + n;
      out[o] = acc[nc][r] + bs + sums[o] * rc8[r];
    }
  }
}

// time_new = time_emb @ W_T.T + b_T   (tiny: 3 MFLOP, scalar f32)
__global__ void time_update_kernel(const float* __restrict__ time_emb,
                                   const float* __restrict__ W_T,
                                   const float* __restrict__ b_T,
                                   float* __restrict__ out_t) {
  int idx = blockIdx.x * blockDim.x + threadIdx.x;
  if (idx >= NTIMES * TIMD) return;
  const int t = idx >> 6, c = idx & 63;
  const float* te = time_emb + t * TIMD;
  const float* w  = W_T + c * TIMD;
  float s = b_T[c];
#pragma unroll
  for (int k = 0; k < TIMD; ++k) s = fmaf(te[k], w[k], s);
  out_t[idx] = s;
}

// ---------------------------------------------------------------- launcher
static inline size_t align_up(size_t x, size_t a) { return (x + a - 1) & ~(a - 1); }

extern "C" void kernel_launch(void* const* d_in, const int* in_sizes, int n_in,
                              void* d_out, int out_size, void* d_ws, size_t ws_size,
                              hipStream_t stream) {
  const float* ent_emb  = (const float*)d_in[0];
  const float* rel_emb  = (const float*)d_in[1];
  const float* time_emb = (const float*)d_in[2];
  const float* W_I = (const float*)d_in[3];
  const float* b_I = (const float*)d_in[4];
  const float* W_O = (const float*)d_in[5];
  const float* b_O = (const float*)d_in[6];
  const float* W_S = (const float*)d_in[7];
  const float* b_S = (const float*)d_in[8];
  const float* W_T = (const float*)d_in[9];
  const float* b_T = (const float*)d_in[10];
  const int* src      = (const int*)d_in[11];
  const int* dst      = (const int*)d_in[12];
  const int* b_rel    = (const int*)d_in[13];
  const int* time_idx = (const int*)d_in[14];
  const int* inv      = (const int*)d_in[15];

  float* out = (float*)d_out;
  char*  wsb = (char*)d_ws;

  // workspace layout
  size_t off = 0;
  float* sums = (float*)(wsb + off); off += (size_t)NNODES * ENT * 4;
  float* cnt  = (float*)(wsb + off); off += (size_t)NNODES * 4;
  off = align_up(off, 32);
  ushort_t* ent16 = (ushort_t*)(wsb + off); off += (size_t)NNODES * ENT * 2;
  off = align_up(off, 32);
  ushort_t* rel16 = (ushort_t*)(wsb + off); off += (size_t)NRELS * RELD * 2;
  off = align_up(off, 32);
  ushort_t* tim16 = (ushort_t*)(wsb + off); off += (size_t)NTIMES * TIMD * 2;
  off = align_up(off, 32);
  ushort_t* wI16  = (ushort_t*)(wsb + off); off += (size_t)ENT * COMP * 2;
  off = align_up(off, 32);
  ushort_t* wO16  = (ushort_t*)(wsb + off); off += (size_t)ENT * COMP * 2;
  off = align_up(off, 32);
  ushort_t* wS16  = (ushort_t*)(wsb + off); off += (size_t)ENT * ENT * 2;

  // 1) zero scatter accumulators
  const int totz = NNODES * ENT + NNODES;
  zero_ws_kernel<<<(totz + 255) / 256, 256, 0, stream>>>(sums, totz);

  // 2) one-time f32 -> bf16 conversions (all sizes divisible by 4)
  auto cvt = [&](const float* s, ushort_t* d, int n) {
    const int n4 = n / 4;
    cvt_f32_bf16_kernel<<<(n4 + 255) / 256, 256, 0, stream>>>(s, d, n4);
  };
  cvt(ent_emb,  ent16, NNODES * ENT);
  cvt(rel_emb,  rel16, NRELS * RELD);
  cvt(time_emb, tim16, NTIMES * TIMD);
  cvt(W_I, wI16, ENT * COMP);
  cvt(W_O, wO16, ENT * COMP);
  cvt(W_S, wS16, ENT * ENT);

  // 3) edge messages: bf16 WMMA + atomic scatter-sum (160 KB dynamic LDS)
  const int etiles  = NEDGES / 16;                   // 31250
  const int eblocks = (etiles + 7) / 8;              // 8 waves / WG
  const size_t esmem = (size_t)2 * ENT * COMP * sizeof(ushort_t); // 163840 B
  edge_msg_kernel<<<eblocks, 256, esmem, stream>>>(
      rel16, ent16, tim16, wI16, wO16, b_I, b_O,
      src, dst, b_rel, time_idx, inv, sums, cnt);

  // 4) node update: WMMA GEMM + mean-agg epilogue
  const int ntiles  = NNODES / 16;                   // 3125
  const int nblocks = (ntiles + 7) / 8;
  node_update_kernel<<<nblocks, 256, 0, stream>>>(
      ent_emb, ent16, wS16, b_S, sums, cnt, out);

  // 5) time update (appended after ent_new)
  const int ttot = NTIMES * TIMD;
  time_update_kernel<<<(ttot + 255) / 256, 256, 0, stream>>>(
      time_emb, W_T, b_T, out + (size_t)NNODES * ENT);
}
